// BiMPM_5669356832704
// MI455X (gfx1250) — compile-verified, matrix operand order checked by reading, not run on hardware
//
#include <hip/hip_runtime.h>
#include <hip/hip_bf16.h>
#include <math.h>
#include <stdint.h>

// ---------------- problem constants ----------------
#define BB 32     // batch
#define SS 64     // sequence length
#define HH 300    // hidden (and embed dim DD)
#define DD 300
#define LL 20     // perspectives
#define CC 2      // classes
#define GG 1200   // 4H gates
#define OO 600    // 2H bilstm output (f32 buffers)
#define OHP 640   // padded f16 bilstm output row (2 x 320)
#define MC 160    // 8L match vector columns
#define KP 320    // K=300 padded to multiple of 32
#define HP 304    // H padded to multiple of 16 (meanmat LDS rows)
#define NP 1216   // 1200 gate rows padded (+16 overhang rows)
#define K1P 1216  // pred K padded
#define EPSF 1e-8f

typedef __attribute__((ext_vector_type(16))) _Float16 v16h;
typedef __attribute__((ext_vector_type(8)))  float    v8f;

// =====================================================================
// Core WMMA K-loop: C += A(16 x 32k) * B(32k x 16), f16 operands already
// padded & 32B-aligned -> unconditional vector loads, no masks.
// rs* in halves (multiples of 16).
// =====================================================================
__device__ __forceinline__ v8f wmma_run(const _Float16* __restrict__ A, int rsA,
                                        const _Float16* __restrict__ Bm, int rsB,
                                        int ksteps, v8f acc)
{
  const int lane = threadIdx.x & 31;
  const int half = lane >> 4, idx = lane & 15;
  const _Float16* ap = A  + (size_t)idx * rsA + half * 16;
  const _Float16* bp = Bm + (size_t)idx * rsB + half * 16;
  for (int s = 0; s < ksteps; ++s) {
    v16h av = *(const v16h*)ap;
    v16h bv = *(const v16h*)bp;
    acc = __builtin_amdgcn_wmma_f32_16x16x32_f16(false, av, false, bv,
                                                 (short)0, acc, false, false);
    ap += 32; bp += 32;
  }
  return acc;
}

// =====================================================================
// K0a: f32 (N,K) weight -> zero-padded f16 (Np,Kp)
// =====================================================================
__global__ void k_w16(const float* __restrict__ src, _Float16* __restrict__ dst,
                      int N, int K, int Np, int Kp)
{
  int gid = blockIdx.x * blockDim.x + threadIdx.x;
  if (gid >= Np * Kp) return;
  int k = gid % Kp, n = gid / Kp;
  float v = (n < N && k < K) ? src[(size_t)n * K + k] : 0.f;
  dst[gid] = (_Float16)v;
}

// K0b: zero an f16 buffer
__global__ void k_zero16(_Float16* __restrict__ p, int n)
{
  int gid = blockIdx.x * blockDim.x + threadIdx.x;
  if (gid < n) p[gid] = (_Float16)0.f;
}

// =====================================================================
// K1: embedding gather -> padded f16 (B*S, KP)
// =====================================================================
__global__ void k_gather16(const float* __restrict__ emb,
                           const int* __restrict__ sa, const int* __restrict__ sb,
                           _Float16* __restrict__ s1h, _Float16* __restrict__ s2h)
{
  int gid = blockIdx.x * blockDim.x + threadIdx.x;
  if (gid >= BB * SS * KP) return;
  int k  = gid % KP;
  int bs = gid / KP;
  float a = (k < DD) ? emb[(size_t)sa[bs] * DD + k] : 0.f;
  float b = (k < DD) ? emb[(size_t)sb[bs] * DD + k] : 0.f;
  s1h[gid] = (_Float16)a;
  s2h[gid] = (_Float16)b;
}

// =====================================================================
// K2: WMMA GEMM + bias: C[M,N] = A16[M,*] * W16[N,*]^T + bias[N]
// The 16-row A strip (contiguous, shared by all 8 waves) is copied into
// LDS with GLOBAL_LOAD_ASYNC_TO_LDS_B128 (ASYNCcnt), then A-fragments are
// LDS vector loads while W-fragments stream from global.
// =====================================================================
__global__ __launch_bounds__(256) void k_gemm_bias(
    const _Float16* __restrict__ A16, int rsA,
    const _Float16* __restrict__ W16, int rsW,
    const float* __restrict__ bias,
    float* __restrict__ C, int M, int N, int ksteps)
{
  __shared__ _Float16 ldsA[16 * K1P] __attribute__((aligned(64)));  // 38 KB max
  const int wave = threadIdx.x >> 5;
  const int row0 = blockIdx.y * 16;
  const int col0 = blockIdx.x * 128 + wave * 16;
  if (row0 >= M) return;                               // block-uniform

  // ---- async cooperative copy of the A strip (16 * rsA halves) ----
  {
    const unsigned ldsBase = (unsigned)(uintptr_t)&ldsA[0];
    const uint64_t gBase   = (uint64_t)(uintptr_t)(A16 + (size_t)row0 * rsA);
    const int chunks = 2 * rsA;                        // 16-byte chunks
    for (int c = threadIdx.x; c < chunks; c += 256) {
      const unsigned lo = ldsBase + (unsigned)c * 16u;
      const uint64_t ga = gBase + (uint64_t)c * 16u;
      asm volatile("global_load_async_to_lds_b128 %0, %1, off"
                   :: "v"(lo), "v"(ga) : "memory");
    }
    asm volatile("s_wait_asynccnt 0x0" ::: "memory");
  }
  __syncthreads();

  if (col0 < N) {                                      // wave-uniform
    v8f acc = {0.f,0.f,0.f,0.f,0.f,0.f,0.f,0.f};
    acc = wmma_run(ldsA, rsA, W16 + (size_t)col0 * rsW, rsW, ksteps, acc);
    const int nRem = (col0 + 16 <= N) ? 16 : (N - col0);
    const int lane = threadIdx.x & 31;
    const int half = lane >> 4, idx = lane & 15;
    if (idx < nRem) {
      const float bz = bias[col0 + idx];
#pragma unroll
      for (int v = 0; v < 8; ++v)
        C[(size_t)(row0 + v + 8 * half) * N + col0 + idx] = acc[v] + bz;
    }
  }
}

// =====================================================================
// K3: LSTM scan. grid.x = 4 combos {s1-fwd, s1-bwd, s2-fwd, s2-bwd}.
// h (f16, padded) and c (f32) live in LDS; Whh pre-converted padded f16
// (NP=1216 rows so the +15 col overhang of wave 18 stays in-bounds).
// Per step: 80 unconditional-load WMMAs per wave.
// =====================================================================
__global__ __launch_bounds__(608) void k_lstm_scan(
    const float* __restrict__ xg0, const float* __restrict__ xg1,
    const float* __restrict__ xg2, const float* __restrict__ xg3,
    const _Float16* __restrict__ Whh16f, const _Float16* __restrict__ Whh16b,
    float* __restrict__ o1, float* __restrict__ o2,       // (B,S,600) f32 or null
    _Float16* __restrict__ o1h, _Float16* __restrict__ o2h, // (B*S,640) f16 or null
    float* __restrict__ hT)                               // (B,1200) or null
{
  __shared__ _Float16 ldsH[BB * KP] __attribute__((aligned(32)));  // 20 KB
  __shared__ float    ldsC[BB * HH];                               // 37.5 KB
  const int combo = blockIdx.x;
  const int rev   = combo & 1;
  const float* xg = (combo == 0) ? xg0 : (combo == 1) ? xg1
                  : (combo == 2) ? xg2 : xg3;
  const _Float16* Whh = rev ? Whh16b : Whh16f;
  float*     o   = (combo >> 1) ? o2  : o1;
  _Float16*  o16 = (combo >> 1) ? o2h : o1h;

  for (int i = threadIdx.x; i < BB * KP; i += blockDim.x) ldsH[i] = (_Float16)0.f;
  for (int i = threadIdx.x; i < BB * HH; i += blockDim.x) ldsC[i] = 0.f;
  __syncthreads();

  const int wave = threadIdx.x >> 5;           // 0..18
  const int lane = threadIdx.x & 31;
  const int half = lane >> 4, idx = lane & 15;
  const int n0   = wave * 16;
  const int nRem = (n0 + 16 <= HH) ? 16 : (HH - n0);
  const int n    = n0 + idx;                   // may overhang to 303 (masked at store)
  const bool nOK = idx < nRem;

  for (int step = 0; step < SS; ++step) {
    const int t = rev ? (SS - 1 - step) : step;
    v8f acc[4][2];
#pragma unroll
    for (int g = 0; g < 4; ++g)
#pragma unroll
      for (int mt = 0; mt < 2; ++mt)
        acc[g][mt] = (v8f){0.f,0.f,0.f,0.f,0.f,0.f,0.f,0.f};

    for (int k0 = 0; k0 < KP; k0 += 32) {
      const int kb = k0 + half * 16;
      v16h a0 = *(const v16h*)&ldsH[(idx)      * KP + kb];
      v16h a1 = *(const v16h*)&ldsH[(16 + idx) * KP + kb];
#pragma unroll
      for (int g = 0; g < 4; ++g) {
        v16h bv = *(const v16h*)(Whh + (size_t)(g * HH + n) * KP + kb);
        acc[g][0] = __builtin_amdgcn_wmma_f32_16x16x32_f16(
            false, a0, false, bv, (short)0, acc[g][0], false, false);
        acc[g][1] = __builtin_amdgcn_wmma_f32_16x16x32_f16(
            false, a1, false, bv, (short)0, acc[g][1], false, false);
      }
    }

    float hnew[2][8];
#pragma unroll
    for (int mt = 0; mt < 2; ++mt) {
#pragma unroll
      for (int v = 0; v < 8; ++v) {
        const int b = mt * 16 + v + 8 * half;          // batch row (M == B == 32)
        float hv = 0.f;
        if (nOK) {
          const float* zrow = xg + (size_t)(b * SS + t) * GG;
          const float zi = zrow[n]          + acc[0][mt][v];
          const float zf = zrow[HH + n]     + acc[1][mt][v];
          const float zg = zrow[2 * HH + n] + acc[2][mt][v];
          const float zo = zrow[3 * HH + n] + acc[3][mt][v];
          const float co = ldsC[b * HH + n];
          const float si = 1.f / (1.f + __expf(-zi));
          const float sf = 1.f / (1.f + __expf(-zf));
          const float so = 1.f / (1.f + __expf(-zo));
          const float cn = sf * co + si * tanhf(zg);
          hv = so * tanhf(cn);
          ldsC[b * HH + n] = cn;
          if (o)   o  [(size_t)(b * SS + t) * OO  + rev * HH  + n] = hv;
          if (o16) o16[(size_t)(b * SS + t) * OHP + rev * KP  + n] = (_Float16)hv;
          if (hT && step == SS - 1) hT[(size_t)b * GG + combo * HH + n] = hv;
        }
        hnew[mt][v] = hv;
      }
    }
    __syncthreads();   // all waves done reading old h
#pragma unroll
    for (int mt = 0; mt < 2; ++mt)
#pragma unroll
      for (int v = 0; v < 8; ++v) {
        const int b = mt * 16 + v + 8 * half;
        if (nOK) ldsH[b * KP + n] = (_Float16)hnew[mt][v];
      }
    __syncthreads();   // new h visible before next step
  }
}

// =====================================================================
// K4: row L2 norms of the four bilstm half-outputs (f32)
// =====================================================================
__global__ void k_norms(const float* __restrict__ o1, const float* __restrict__ o2,
                        float* __restrict__ ns1f, float* __restrict__ ns1b,
                        float* __restrict__ ns2f, float* __restrict__ ns2b)
{
  int gid = blockIdx.x * blockDim.x + threadIdx.x;
  if (gid >= BB * SS * 4) return;
  int kind = gid / (BB * SS);
  int bs   = gid % (BB * SS);
  const float* src = (kind < 2 ? o1 : o2) + (size_t)bs * OO + ((kind & 1) ? HH : 0);
  float s = 0.f;
  for (int h = 0; h < HH; ++h) { float v = src[h]; s += v * v; }
  float* dst = (kind == 0) ? ns1f : (kind == 1) ? ns1b : (kind == 2) ? ns2f : ns2b;
  dst[bs] = sqrtf(s);
}

// =====================================================================
// K5: pairwise match (WMMA): per (b,l,z) a 64x64x300 GEMM.
// W^2-scaled A staged once into LDS f16; B read from padded f16 o2h.
// z=0: (s1f, s2b, W_maxpool_f)   z=1: (s1b, s2b, W_maxpool_b)
// =====================================================================
__global__ __launch_bounds__(256) void k_pairwise(
    const float* __restrict__ o1, const _Float16* __restrict__ o2h,
    const float* __restrict__ Wmf, const float* __restrict__ Wmb,
    const float* __restrict__ ns1f, const float* __restrict__ ns1b,
    const float* __restrict__ ns2b,
    float* __restrict__ pf, float* __restrict__ pb)
{
  __shared__ _Float16 ldsA[SS * KP] __attribute__((aligned(32)));  // 40 KB
  const int b = blockIdx.x, l = blockIdx.y, z = blockIdx.z;
  const float* a32  = o1 + (size_t)b * SS * OO + (z ? HH : 0);
  const float* wrow = (z ? Wmb : Wmf) + l * HH;
  const float* n1   = (z ? ns1b : ns1f) + b * SS;
  const float* n2   = ns2b + b * SS;
  float* outp = (z ? pb : pf) + (size_t)b * SS * SS * LL;

  for (int e = threadIdx.x; e < SS * KP; e += 256) {
    const int i = e / KP, k = e % KP;
    float v = 0.f;
    if (k < HH) { const float w = wrow[k]; v = a32[(size_t)i * OO + k] * w * w; }
    ldsA[e] = (_Float16)v;
  }
  __syncthreads();

  const int wave = threadIdx.x >> 5, lane = threadIdx.x & 31;
  const int half = lane >> 4, idx = lane & 15;
  const _Float16* Bm = o2h + (size_t)b * SS * OHP + KP;   // s2b half, padded
  for (int r = 0; r < 2; ++r) {
    const int tile = wave * 2 + r;       // 16 tiles = 4x4 over 64x64
    const int it = tile >> 2, jt = tile & 3;
    v8f acc = {0.f,0.f,0.f,0.f,0.f,0.f,0.f,0.f};
    acc = wmma_run(ldsA + (size_t)(it * 16) * KP, KP,
                   Bm + (size_t)(jt * 16) * OHP, OHP, KP / 32, acc);
    const int j = jt * 16 + idx;
#pragma unroll
    for (int v = 0; v < 8; ++v) {
      const int i = it * 16 + v + 8 * half;
      outp[((size_t)i * SS + j) * LL + l] = acc[v] / (n1[i] * n2[j]);
    }
  }
}

// =====================================================================
// K6: cosine matrices (WMMA) straight from padded f16 o1h/o2h
// =====================================================================
__global__ __launch_bounds__(256) void k_cosine(
    const _Float16* __restrict__ o1h, const _Float16* __restrict__ o2h,
    const float* __restrict__ ns1f, const float* __restrict__ ns1b,
    const float* __restrict__ ns2f, const float* __restrict__ ns2b,
    float* __restrict__ cf, float* __restrict__ cb)
{
  const int b = blockIdx.x, z = blockIdx.y;
  const _Float16* A  = o1h + (size_t)b * SS * OHP + z * KP;
  const _Float16* Bm = o2h + (size_t)b * SS * OHP + z * KP;
  const float* n1 = (z ? ns1b : ns1f) + b * SS;
  const float* n2 = (z ? ns2b : ns2f) + b * SS;
  float* outp = (z ? cb : cf) + (size_t)b * SS * SS;
  const int wave = threadIdx.x >> 5, lane = threadIdx.x & 31;
  const int half = lane >> 4, idx = lane & 15;
  for (int r = 0; r < 2; ++r) {
    const int tile = wave * 2 + r;
    const int it = tile >> 2, jt = tile & 3;
    v8f acc = {0.f,0.f,0.f,0.f,0.f,0.f,0.f,0.f};
    acc = wmma_run(A + (size_t)(it * 16) * OHP, OHP,
                   Bm + (size_t)(jt * 16) * OHP, OHP, KP / 32, acc);
    const int j = jt * 16 + idx;
#pragma unroll
    for (int v = 0; v < 8; ++v) {
      const int i = it * 16 + v + 8 * half;
      outp[(size_t)i * SS + j] = acc[v] / (n1[i] * n2[j]);
    }
  }
}

// =====================================================================
// K7: cosine row/col sums
// =====================================================================
__global__ void k_cos_sums(const float* __restrict__ cf, const float* __restrict__ cb,
                           float* __restrict__ rsf, float* __restrict__ csf,
                           float* __restrict__ rsb, float* __restrict__ csb)
{
  int gid = blockIdx.x * blockDim.x + threadIdx.x;
  if (gid >= BB * SS * 4) return;
  int kind = gid / (BB * SS);
  int bs = gid % (BB * SS);
  int b = bs / SS, s = bs % SS;
  const float* cm = ((kind < 2) ? cf : cb) + (size_t)b * SS * SS;
  float acc = 0.f;
  if ((kind & 1) == 0) { for (int i = 0; i < SS; ++i) acc += cm[i * SS + s]; }
  else                 { for (int j = 0; j < SS; ++j) acc += cm[s * SS + j]; }
  float* dst = (kind == 0) ? rsf : (kind == 1) ? csf : (kind == 2) ? rsb : csb;
  dst[bs] = acc;
}

// =====================================================================
// K8: attentive means (WMMA, K=64): LDS-staged cos (opt. transposed) and
// transposed s^T so all fragment loads are unconditional ds vectors.
// c0: mean_s1_f = cf @ s2f / csum_f   c1: mean_s1_b = cb @ s2f / csum_b
// c2: mean_s2_f = cf^T @ s1f / rsum_f c3: mean_s2_b = cb^T @ s1f / rsum_b
// =====================================================================
__global__ __launch_bounds__(256) void k_meanmat(
    const float* __restrict__ cf, const float* __restrict__ cb,
    const float* __restrict__ o1, const float* __restrict__ o2,
    const float* __restrict__ rsf, const float* __restrict__ csf,
    const float* __restrict__ rsb, const float* __restrict__ csb,
    float* __restrict__ m1f, float* __restrict__ m1b,
    float* __restrict__ m2f, float* __restrict__ m2b)
{
  __shared__ _Float16 ldsA [SS * SS] __attribute__((aligned(32)));  // 8 KB
  __shared__ _Float16 ldsBT[HP * SS] __attribute__((aligned(32)));  // 38 KB
  const int b = blockIdx.x, c = blockIdx.y;
  const bool trans = (c >= 2);
  const float* cm  = (((c & 1) == 0) ? cf : cb) + (size_t)b * SS * SS;
  const float* src = (trans ? o1 : o2) + (size_t)b * SS * OO;       // s1f / s2f
  const float* divv = ((c == 0) ? csf : (c == 1) ? csb : (c == 2) ? rsf : rsb) + b * SS;
  float* outp = ((c == 0) ? m1f : (c == 1) ? m1b : (c == 2) ? m2f : m2b)
              + (size_t)b * SS * HH;

  for (int e = threadIdx.x; e < SS * SS; e += 256) {
    const int i = e >> 6, j = e & 63;
    const float v = cm[i * SS + j];
    if (trans) ldsA[j * SS + i] = (_Float16)v;
    else       ldsA[i * SS + j] = (_Float16)v;
  }
  for (int e = threadIdx.x; e < HP * SS; e += 256) {
    const int h = e / SS, j = e % SS;
    const float v = (h < HH) ? src[(size_t)j * OO + h] : 0.f;
    ldsBT[h * SS + j] = (_Float16)v;
  }
  __syncthreads();

  const int wave = threadIdx.x >> 5, lane = threadIdx.x & 31;
  const int half = lane >> 4, idx = lane & 15;
  for (int tile = wave; tile < 4 * 19; tile += 8) {
    const int mt = tile / 19, nt = tile % 19;
    const int nRem = ((nt + 1) * 16 <= HH) ? 16 : (HH - nt * 16);
    v8f acc = {0.f,0.f,0.f,0.f,0.f,0.f,0.f,0.f};
    acc = wmma_run(ldsA + (size_t)(mt * 16) * SS, SS,
                   ldsBT + (size_t)(nt * 16) * SS, SS, SS / 32, acc);
    const int col = nt * 16 + idx;
    if (idx < nRem) {
#pragma unroll
      for (int v = 0; v < 8; ++v) {
        const int row = mt * 16 + v + 8 * half;
        outp[(size_t)row * HH + col] = acc[v] / divv[row];
      }
    }
  }
}

// =====================================================================
// K9: max-attentive vectors (VALU reductions over S)
// =====================================================================
__global__ void k_maxattn(const float* __restrict__ o1, const float* __restrict__ o2,
                          const float* __restrict__ cf, const float* __restrict__ cb,
                          float* __restrict__ x1f, float* __restrict__ x1b,
                          float* __restrict__ x2f, float* __restrict__ x2b)
{
  int gid = blockIdx.x * blockDim.x + threadIdx.x;
  if (gid >= BB * SS * HH) return;
  const int c = blockIdx.y;
  const int h = gid % HH;
  const int s = (gid / HH) % SS;
  const int b = gid / (HH * SS);
  const float* cm = (((c & 1) == 0) ? cf : cb) + (size_t)b * SS * SS;
  float m = -INFINITY;
  if (c < 2) {
    for (int j = 0; j < SS; ++j)
      m = fmaxf(m, o2[(size_t)(b * SS + j) * OO + h] * cm[s * SS + j]);
  } else {
    for (int i = 0; i < SS; ++i)
      m = fmaxf(m, o1[(size_t)(b * SS + i) * OO + h] * cm[i * SS + s]);
  }
  float* dst = (c == 0) ? x1f : (c == 1) ? x1b : (c == 2) ? x2f : x2b;
  dst[(size_t)(b * SS + s) * HH + h] = m;
}

// =====================================================================
// K10: maxpool over pairwise tensor -> v1/v2 columns 20..39 / 100..119
// =====================================================================
__global__ void k_maxpool(const float* __restrict__ pf, const float* __restrict__ pb,
                          float* __restrict__ v1, float* __restrict__ v2)
{
  int gid = blockIdx.x * blockDim.x + threadIdx.x;
  if (gid >= BB * SS * LL) return;
  const int c = blockIdx.y;
  const int l = gid % LL;
  const int s = (gid / LL) % SS;
  const int b = gid / (LL * SS);
  const float* p = (((c & 1) == 0) ? pf : pb) + (size_t)b * SS * SS * LL;
  float m = -INFINITY;
  if (c < 2) { for (int j = 0; j < SS; ++j) m = fmaxf(m, p[((size_t)s * SS + j) * LL + l]); }
  else       { for (int i = 0; i < SS; ++i) m = fmaxf(m, p[((size_t)i * SS + s) * LL + l]); }
  float* dst = (c < 2) ? v1 : v2;
  const int colOff = ((c & 1) == 0) ? 20 : 100;
  dst[(size_t)(b * SS + s) * MC + colOff + l] = m;
}

// =====================================================================
// K11: generic full-match -> writes L columns into v1/v2 at colOff
// =====================================================================
__global__ void k_full_match(const float* __restrict__ a_base, int a_rs, int a_bs,
                             const float* __restrict__ c_base, int c_rs, int c_bs, int c3d,
                             const float* __restrict__ Wm,
                             float* __restrict__ outb, int colOff)
{
  int gid = blockIdx.x * blockDim.x + threadIdx.x;
  if (gid >= BB * SS * LL) return;
  const int l = gid % LL;
  const int s = (gid / LL) % SS;
  const int b = gid / (LL * SS);
  const float* a = a_base + (size_t)b * a_bs + (size_t)s * a_rs;
  const float* c = c3d ? (c_base + (size_t)b * c_bs + (size_t)s * c_rs)
                       : (c_base + (size_t)b * c_bs);
  const float* w = Wm + l * HH;
  float num = 0.f, na = 0.f, nc = 0.f;
  for (int h = 0; h < HH; ++h) {
    float ww = w[h]; ww *= ww;
    const float av = a[h], cv = c[h];
    num += ww * av * cv;
    na  += ww * av * av;
    nc  += ww * cv * cv;
  }
  const float d1 = fmaxf(sqrtf(na), EPSF);
  const float d2 = fmaxf(sqrtf(nc), EPSF);
  outb[(size_t)(b * SS + s) * MC + colOff + l] = num / (d1 * d2);
}

// =====================================================================
// K12: prediction tail: x=tanh(pre1); logits=x@W2^T+b2; log_softmax
// =====================================================================
__global__ void k_pred_final(const float* __restrict__ pre1,
                             const float* __restrict__ W2p, const float* __restrict__ b2p,
                             float* __restrict__ out)
{
  int b = blockIdx.x * blockDim.x + threadIdx.x;
  if (b >= BB) return;
  float l0 = b2p[0], l1 = b2p[1];
  for (int j = 0; j < OO; ++j) {
    const float xv = tanhf(pre1[(size_t)b * OO + j]);
    l0 += xv * W2p[j];
    l1 += xv * W2p[OO + j];
  }
  const float m = fmaxf(l0, l1);
  const float lse = m + logf(__expf(l0 - m) + __expf(l1 - m));
  out[b * CC + 0] = l0 - lse;
  out[b * CC + 1] = l1 - lse;
}

// =====================================================================
// host launcher
// =====================================================================
extern "C" void kernel_launch(void* const* d_in, const int* in_sizes, int n_in,
                              void* d_out, int out_size, void* d_ws, size_t ws_size,
                              hipStream_t stream)
{
  const float* emb       = (const float*)d_in[0];
  const float* ctx_Wih_f = (const float*)d_in[1];
  const float* ctx_Whh_f = (const float*)d_in[2];
  const float* ctx_b_f   = (const float*)d_in[3];
  const float* ctx_Wih_b = (const float*)d_in[4];
  const float* ctx_Whh_b = (const float*)d_in[5];
  const float* ctx_b_b   = (const float*)d_in[6];
  const float* agg_Wih_f = (const float*)d_in[7];
  const float* agg_Whh_f = (const float*)d_in[8];
  const float* agg_b_f   = (const float*)d_in[9];
  const float* agg_Wih_b = (const float*)d_in[10];
  const float* agg_Whh_b = (const float*)d_in[11];
  const float* agg_b_b   = (const float*)d_in[12];
  const float* W_full_f  = (const float*)d_in[13];
  const float* W_full_b  = (const float*)d_in[14];
  const float* W_mp_f    = (const float*)d_in[15];
  const float* W_mp_b    = (const float*)d_in[16];
  const float* W_attn_f  = (const float*)d_in[17];
  const float* W_mattn_f = (const float*)d_in[18];
  const float* pred_W1   = (const float*)d_in[19];
  const float* pred_b1   = (const float*)d_in[20];
  const float* pred_W2   = (const float*)d_in[21];
  const float* pred_b2   = (const float*)d_in[22];
  const int*   sent_a    = (const int*)d_in[23];
  const int*   sent_b    = (const int*)d_in[24];
  float* out = (float*)d_out;

  // ---- workspace carve-up: f32 region then 32B-aligned f16 region ----
  float* wsf = (float*)d_ws;
  size_t off = 0;
  auto allocf = [&](size_t n) { float* p = wsf + off; off += n; return p; };
  float* xg0  = allocf((size_t)BB * SS * GG);
  float* xg1  = allocf((size_t)BB * SS * GG);
  float* xg2  = allocf((size_t)BB * SS * GG);
  float* xg3  = allocf((size_t)BB * SS * GG);
  float* o1   = allocf((size_t)BB * SS * OO);
  float* o2   = allocf((size_t)BB * SS * OO);
  float* pf   = allocf((size_t)BB * SS * SS * LL);
  float* pb   = allocf((size_t)BB * SS * SS * LL);
  float* cf   = allocf((size_t)BB * SS * SS);
  float* cb   = allocf((size_t)BB * SS * SS);
  float* rsf  = allocf((size_t)BB * SS);
  float* csf  = allocf((size_t)BB * SS);
  float* rsb  = allocf((size_t)BB * SS);
  float* csb  = allocf((size_t)BB * SS);
  float* ns1f = allocf((size_t)BB * SS);
  float* ns1b = allocf((size_t)BB * SS);
  float* ns2f = allocf((size_t)BB * SS);
  float* ns2b = allocf((size_t)BB * SS);
  float* m1f  = allocf((size_t)BB * SS * HH);
  float* m1b  = allocf((size_t)BB * SS * HH);
  float* m2f  = allocf((size_t)BB * SS * HH);
  float* m2b  = allocf((size_t)BB * SS * HH);
  float* x1f  = allocf((size_t)BB * SS * HH);
  float* x1b  = allocf((size_t)BB * SS * HH);
  float* x2f  = allocf((size_t)BB * SS * HH);
  float* x2b  = allocf((size_t)BB * SS * HH);
  float* v1   = allocf((size_t)BB * SS * MC);
  float* v2   = allocf((size_t)BB * SS * MC);
  float* mv   = allocf((size_t)BB * GG);
  float* pre1 = allocf((size_t)BB * OO);
  off = (off + 7) & ~(size_t)7;                       // 32-byte align
  _Float16* wsh = (_Float16*)(wsf + off);
  size_t offh = 0;
  auto alloch = [&](size_t n) { _Float16* p = wsh + offh; offh += n; return p; };
  _Float16* s1h   = alloch((size_t)BB * SS * KP);
  _Float16* s2h   = alloch((size_t)BB * SS * KP);
  _Float16* cWihF = alloch((size_t)NP * KP);
  _Float16* cWihB = alloch((size_t)NP * KP);
  _Float16* cWhhF = alloch((size_t)NP * KP);
  _Float16* cWhhB = alloch((size_t)NP * KP);
  _Float16* aWihF = alloch((size_t)NP * MC);
  _Float16* aWihB = alloch((size_t)NP * MC);
  _Float16* aWhhF = alloch((size_t)NP * KP);
  _Float16* aWhhB = alloch((size_t)NP * KP);
  _Float16* pW1h  = alloch((size_t)608 * K1P);
  _Float16* o1h   = alloch((size_t)BB * SS * OHP);
  _Float16* o2h   = alloch((size_t)BB * SS * OHP);
  _Float16* v1h   = alloch((size_t)BB * SS * MC);
  _Float16* v2h   = alloch((size_t)BB * SS * MC);
  _Float16* mvh   = alloch((size_t)BB * K1P);
  (void)ws_size; (void)n_in; (void)in_sizes; (void)out_size;

  auto cdiv = [](int a, int b) { return (a + b - 1) / b; };

  // ---- 0) weight/activation conversions to padded f16 ----
  k_w16<<<cdiv(NP * KP, 256), 256, 0, stream>>>(ctx_Wih_f, cWihF, GG, HH, NP, KP);
  k_w16<<<cdiv(NP * KP, 256), 256, 0, stream>>>(ctx_Wih_b, cWihB, GG, HH, NP, KP);
  k_w16<<<cdiv(NP * KP, 256), 256, 0, stream>>>(ctx_Whh_f, cWhhF, GG, HH, NP, KP);
  k_w16<<<cdiv(NP * KP, 256), 256, 0, stream>>>(ctx_Whh_b, cWhhB, GG, HH, NP, KP);
  k_w16<<<cdiv(NP * MC, 256), 256, 0, stream>>>(agg_Wih_f, aWihF, GG, MC, NP, MC);
  k_w16<<<cdiv(NP * MC, 256), 256, 0, stream>>>(agg_Wih_b, aWihB, GG, MC, NP, MC);
  k_w16<<<cdiv(NP * KP, 256), 256, 0, stream>>>(agg_Whh_f, aWhhF, GG, HH, NP, KP);
  k_w16<<<cdiv(NP * KP, 256), 256, 0, stream>>>(agg_Whh_b, aWhhB, GG, HH, NP, KP);
  k_w16<<<cdiv(608 * K1P, 256), 256, 0, stream>>>(pred_W1, pW1h, OO, GG, 608, K1P);
  k_zero16<<<cdiv(BB * SS * OHP, 256), 256, 0, stream>>>(o1h, BB * SS * OHP);
  k_zero16<<<cdiv(BB * SS * OHP, 256), 256, 0, stream>>>(o2h, BB * SS * OHP);

  // ---- 1) gather embeddings -> padded f16 ----
  k_gather16<<<cdiv(BB * SS * KP, 256), 256, 0, stream>>>(emb, sent_a, sent_b, s1h, s2h);

  // ---- 2) context xg = x @ Wih^T + b (4 WMMA GEMMs) ----
  dim3 gC(cdiv(GG, 128), cdiv(BB * SS, 16));
  k_gemm_bias<<<gC, 256, 0, stream>>>(s1h, KP, cWihF, KP, ctx_b_f, xg0, BB * SS, GG, KP / 32);
  k_gemm_bias<<<gC, 256, 0, stream>>>(s1h, KP, cWihB, KP, ctx_b_b, xg1, BB * SS, GG, KP / 32);
  k_gemm_bias<<<gC, 256, 0, stream>>>(s2h, KP, cWihF, KP, ctx_b_f, xg2, BB * SS, GG, KP / 32);
  k_gemm_bias<<<gC, 256, 0, stream>>>(s2h, KP, cWihB, KP, ctx_b_b, xg3, BB * SS, GG, KP / 32);

  // ---- 3) context bilstm scan ----
  k_lstm_scan<<<4, 608, 0, stream>>>(xg0, xg1, xg2, xg3, cWhhF, cWhhB,
                                     o1, o2, o1h, o2h, nullptr);

  // ---- 4) norms ----
  k_norms<<<cdiv(BB * SS * 4, 256), 256, 0, stream>>>(o1, o2, ns1f, ns1b, ns2f, ns2b);

  // ---- 5) pairwise match (WMMA) + maxpool ----
  k_pairwise<<<dim3(BB, LL, 2), 256, 0, stream>>>(o1, o2h, W_mp_f, W_mp_b,
                                                  ns1f, ns1b, ns2b, pf, pb);
  k_maxpool<<<dim3(cdiv(BB * SS * LL, 256), 4), 256, 0, stream>>>(pf, pb, v1, v2);

  // ---- 6) cosine matrices (WMMA) + sums ----
  k_cosine<<<dim3(BB, 2), 256, 0, stream>>>(o1h, o2h, ns1f, ns1b, ns2f, ns2b, cf, cb);
  k_cos_sums<<<cdiv(BB * SS * 4, 256), 256, 0, stream>>>(cf, cb, rsf, csf, rsb, csb);

  // ---- 7) attentive means (WMMA) and max-attentive ----
  k_meanmat<<<dim3(BB, 4), 256, 0, stream>>>(cf, cb, o1, o2, rsf, csf, rsb, csb,
                                             m1f, m1b, m2f, m2b);
  k_maxattn<<<dim3(cdiv(BB * SS * HH, 256), 4), 256, 0, stream>>>(o1, o2, cf, cb,
                                                                  x1f, x1b, x2f, x2b);

  // ---- 8) full-match family -> v1/v2 columns ----
  const int fmG = cdiv(BB * SS * LL, 256);
  const int oBS = SS * OO, hBS = SS * HH;
  k_full_match<<<fmG, 256, 0, stream>>>(o1,      OO, oBS, o2 + (SS - 1) * OO, 0, oBS, 0, W_full_f,  v1, 0);
  k_full_match<<<fmG, 256, 0, stream>>>(o1 + HH, OO, oBS, o2 + HH,            0, oBS, 0, W_full_b,  v1, 80);
  k_full_match<<<fmG, 256, 0, stream>>>(o1,      OO, oBS, m1f, HH, hBS, 1,           W_attn_f,  v1, 40);
  k_full_match<<<fmG, 256, 0, stream>>>(o1 + HH, OO, oBS, m1b, HH, hBS, 1,           W_attn_f,  v1, 120);
  k_full_match<<<fmG, 256, 0, stream>>>(o1,      OO, oBS, x1f, HH, hBS, 1,           W_mattn_f, v1, 60);
  k_full_match<<<fmG, 256, 0, stream>>>(o1 + HH, OO, oBS, x1b, HH, hBS, 1,           W_mattn_f, v1, 140);
  k_full_match<<<fmG, 256, 0, stream>>>(o2,      OO, oBS, o1 + (SS - 1) * OO, 0, oBS, 0, W_full_f,  v2, 0);
  k_full_match<<<fmG, 256, 0, stream>>>(o2 + HH, OO, oBS, o1 + HH,            0, oBS, 0, W_full_b,  v2, 80);
  k_full_match<<<fmG, 256, 0, stream>>>(o2,      OO, oBS, m2f, HH, hBS, 1,           W_attn_f,  v2, 40);
  k_full_match<<<fmG, 256, 0, stream>>>(o2 + HH, OO, oBS, m2b, HH, hBS, 1,           W_attn_f,  v2, 120);
  k_full_match<<<fmG, 256, 0, stream>>>(o2,      OO, oBS, x2f, HH, hBS, 1,           W_mattn_f, v2, 60);
  k_full_match<<<fmG, 256, 0, stream>>>(o2 + HH, OO, oBS, x2b, HH, hBS, 1,           W_mattn_f, v2, 140);

  // ---- 9) aggregation: convert v1/v2, xg GEMMs (K=160), scan -> hT in mv ----
  k_w16<<<cdiv(BB * SS * MC, 256), 256, 0, stream>>>(v1, v1h, BB * SS, MC, BB * SS, MC);
  k_w16<<<cdiv(BB * SS * MC, 256), 256, 0, stream>>>(v2, v2h, BB * SS, MC, BB * SS, MC);
  k_gemm_bias<<<gC, 256, 0, stream>>>(v1h, MC, aWihF, MC, agg_b_f, xg0, BB * SS, GG, MC / 32);
  k_gemm_bias<<<gC, 256, 0, stream>>>(v1h, MC, aWihB, MC, agg_b_b, xg1, BB * SS, GG, MC / 32);
  k_gemm_bias<<<gC, 256, 0, stream>>>(v2h, MC, aWihF, MC, agg_b_f, xg2, BB * SS, GG, MC / 32);
  k_gemm_bias<<<gC, 256, 0, stream>>>(v2h, MC, aWihB, MC, agg_b_b, xg3, BB * SS, GG, MC / 32);
  k_lstm_scan<<<4, 608, 0, stream>>>(xg0, xg1, xg2, xg3, aWhhF, aWhhB,
                                     nullptr, nullptr, nullptr, nullptr, mv);

  // ---- 10) prediction head ----
  k_w16<<<cdiv(BB * K1P, 256), 256, 0, stream>>>(mv, mvh, BB, GG, BB, K1P);
  dim3 gP(cdiv(OO, 128), cdiv(BB, 16));
  k_gemm_bias<<<gP, 256, 0, stream>>>(mvh, K1P, pW1h, K1P, pred_b1, pre1, BB, OO, K1P / 32);
  k_pred_final<<<1, 64, 0, stream>>>(pre1, pred_W2, pred_b2, out);
}
